// MambaLayer_137438954035
// MI455X (gfx1250) — matrix operand order repaired and checked by reading, not compiled, not run on hardware
//
#include <hip/hip_runtime.h>
#include <hip/hip_bf16.h>
#include <math.h>

#define BATCH 2
#define SEQL  2048
#define EDIM  1024
#define NST   16
#define DC    4
#define DI    2048   // EDIM * EXP

typedef __attribute__((ext_vector_type(16))) __bf16 vbf16;
typedef __attribute__((ext_vector_type(8)))  __bf16 vbf8;
typedef __attribute__((ext_vector_type(8)))  float  v8f;

__device__ __forceinline__ unsigned short f2bf_rne(float f) {
  unsigned int u = __float_as_uint(f);
  unsigned int r = 0x7FFFu + ((u >> 16) & 1u);
  u += r;
  return (unsigned short)(u >> 16);
}

// ---------------------------------------------------------------- fp32 -> bf16
__global__ void k_f2bf(const float* __restrict__ src,
                       unsigned short* __restrict__ dst, long long n) {
  long long i = (long long)blockIdx.x * blockDim.x + threadIdx.x;
  long long stride = (long long)gridDim.x * blockDim.x;
  for (; i < n; i += stride) dst[i] = f2bf_rne(src[i]);
}

// ---------------------------------------------------------------- WMMA GEMM
// C[M,N] = A[M,K] (bf16, row-major) * B[N,K]^T (bf16, row-major rows = N)
// block = 256 threads = 8 waves; each wave computes a 64x64 macro-tile
// (4x4 grid of 16x16 WMMA tiles -> 16 WMMAs per K-step sharing 4 A-frags
// and 4 B-frags). Waves arranged 2(M) x 4(N): block tile = 128(M) x 256(N).
// Requires M%128==0, N%256==0, K%32==0.
__global__ __launch_bounds__(256)
void k_gemm_bf16(const unsigned short* __restrict__ A,
                 const unsigned short* __restrict__ B,
                 float* __restrict__ C,
                 int M, int N, int K) {
  const int lane = threadIdx.x & 31;
  const int wave = threadIdx.x >> 5;   // 0..7
  const int wm = wave & 1;             // 2 waves along M
  const int wn = wave >> 1;            // 4 waves along N
  const int row = lane & 15;
  const int kh  = lane >> 4;           // which K-half this lane holds

  const int m0 = blockIdx.x * 128 + wm * 64;
  const int n0 = blockIdx.y * 256 + wn * 64;

  v8f acc[4][4];
#pragma unroll
  for (int i = 0; i < 4; ++i)
#pragma unroll
    for (int j = 0; j < 4; ++j) acc[i][j] = (v8f){};

  const unsigned short* Arow[4];
  const unsigned short* Brow[4];
#pragma unroll
  for (int i = 0; i < 4; ++i) {
    Arow[i] = A + (size_t)(m0 + 16 * i + row) * K;
    Brow[i] = B + (size_t)(n0 + 16 * i + row) * K;
  }

  for (int k = 0; k < K; k += 32) {
    vbf16 af[4], bf[4];
#pragma unroll
    for (int i = 0; i < 4; ++i) {
      union { vbf8 h[2]; vbf16 v; } u;
      // A frag: elems 0..7 = K(kh*8..+7), elems 8..15 = K(16+kh*8..+7)
      u.h[0] = *(const vbf8*)(Arow[i] + k + kh * 8);
      u.h[1] = *(const vbf8*)(Arow[i] + k + 16 + kh * 8);
      af[i] = u.v;
      // B frag: elems 0..7 = K(kh*8..+7), elems 8..15 = K(16+kh*8..+7), column = lane&15
      u.h[0] = *(const vbf8*)(Brow[i] + k + kh * 8);
      u.h[1] = *(const vbf8*)(Brow[i] + k + 16 + kh * 8);
      bf[i] = u.v;
    }
    if (k + 128 < K) {
      // pull the next panels toward the WGP (lowers to global_prefetch_b8)
      __builtin_prefetch(Arow[0] + k + 128, 0, 1);
      __builtin_prefetch(Brow[0] + k + 128, 0, 1);
    }
#pragma unroll
    for (int i = 0; i < 4; ++i)
#pragma unroll
      for (int j = 0; j < 4; ++j)
        acc[i][j] = __builtin_amdgcn_wmma_f32_16x16x32_bf16(
            false, af[i], false, bf[j], (short)0, acc[i][j], false, false);
  }

#pragma unroll
  for (int i = 0; i < 4; ++i) {
#pragma unroll
    for (int j = 0; j < 4; ++j) {
      const int cn = n0 + 16 * j + row;
#pragma unroll
      for (int r = 0; r < 8; ++r) {
        const int mA = m0 + 16 * i + r + 8 * kh;
        C[(size_t)mA * N + cn] = acc[i][j][r];
      }
    }
  }
}

// ------------------------------------------------- depthwise causal conv + silu
// xz: (B*L, 2*DI) fp32; out xs: (B*L, DI) fp32
__global__ void k_conv_silu(const float* __restrict__ xz,
                            const float* __restrict__ conv_w,  // (DI,1,DC)
                            const float* __restrict__ conv_b,  // (DI)
                            float* __restrict__ xs) {
  long long i = (long long)blockIdx.x * blockDim.x + threadIdx.x;
  const long long total = (long long)BATCH * SEQL * DI;
  if (i >= total) return;
  const int d = (int)(i % DI);
  const long long bl = i / DI;
  const int t = (int)(bl % SEQL);
  const long long brow = bl - t;  // b*L
  float acc = conv_b[d];
#pragma unroll
  for (int j = 0; j < DC; ++j) {
    const int tt = t + j - (DC - 1);
    if (tt >= 0)
      acc += xz[(brow + tt) * (size_t)(2 * DI) + d] * conv_w[d * DC + j];
  }
  xs[bl * (size_t)DI + d] = acc * (1.0f / (1.0f + expf(-acc)));
}

// ------------------------------------------------- ssm projection (N = 33)
// ssm[row, n] = dot(xs[row,:], x_proj_w[n,:]), K = DI
__global__ __launch_bounds__(256)
void k_xproj(const float* __restrict__ xs,
             const float* __restrict__ xpw,  // (33, DI)
             float* __restrict__ ssm) {      // (B*L, 33)
  __shared__ float srow[DI];
  const size_t rowi = blockIdx.x;  // 0..B*L-1
  const float* r = xs + rowi * DI;
  for (int k = threadIdx.x; k < DI; k += 256) srow[k] = r[k];
  __syncthreads();
  const int n = threadIdx.x;
  if (n < 2 * NST + 1) {
    const float* w = xpw + (size_t)n * DI;
    float acc = 0.f;
    for (int k = 0; k < DI; ++k) acc += srow[k] * w[k];
    ssm[rowi * (2 * NST + 1) + n] = acc;
  }
}

// ------------------------------------------------- selective scan + gating
// one block = 16 channels of one batch; thread = (channel, state n)
// writes y * silu(z) as bf16 for the out-projection WMMA GEMM
__global__ __launch_bounds__(256)
void k_scan(const float* __restrict__ ssm,   // (B*L, 33)
            const float* __restrict__ xs,    // (B*L, DI)  (= u)
            const float* __restrict__ xz,    // (B*L, 2*DI); z at col DI+d
            const float* __restrict__ dt_w,  // (DI)
            const float* __restrict__ dt_b,  // (DI)
            const float* __restrict__ A_log, // (DI, NST)
            const float* __restrict__ D_param, // (DI)
            unsigned short* __restrict__ ybf) { // (B*L, DI) bf16
  const int tid = threadIdx.x;
  const int n  = tid & 15;
  const int ch = tid >> 4;                 // 0..15
  const int d  = blockIdx.x * 16 + ch;
  const int b  = blockIdx.y;

  const float Adn = -expf(A_log[(size_t)d * NST + n]);
  const float dtw = dt_w[d];
  const float dtb = dt_b[d];
  const float Dd  = D_param[d];

  float h = 0.f;
  const size_t base = (size_t)b * SEQL;
  for (int l = 0; l < SEQL; ++l) {
    const size_t rl = base + l;
    const float* srow = ssm + rl * (2 * NST + 1);
    const float dtr = srow[0];
    const float Bt  = srow[1 + n];
    const float Ct  = srow[1 + NST + n];
    const float u   = xs[rl * DI + d];
    const float z   = xz[rl * (size_t)(2 * DI) + DI + d];

    const float xdt = dtr * dtw + dtb;
    const float dt  = (xdt > 20.f) ? xdt : log1pf(expf(xdt));  // softplus
    const float dA  = expf(dt * Adn);
    h = dA * h + (dt * u) * Bt;
    float p = h * Ct;
    // reduce over the 16 state lanes of this channel (wave32)
    p += __shfl_xor(p, 1, 32);
    p += __shfl_xor(p, 2, 32);
    p += __shfl_xor(p, 4, 32);
    p += __shfl_xor(p, 8, 32);
    if (n == 0) {
      const float y  = p + Dd * u;
      const float zs = z * (1.0f / (1.0f + expf(-z)));
      ybf[rl * DI + d] = f2bf_rne(y * zs);
    }
  }
}

// ----------------------------------------------------------------------------
extern "C" void kernel_launch(void* const* d_in, const int* in_sizes, int n_in,
                              void* d_out, int out_size, void* d_ws, size_t ws_size,
                              hipStream_t stream) {
  const float* x          = (const float*)d_in[0];
  const float* in_proj_w  = (const float*)d_in[1];  // (2*DI, EDIM)
  const float* conv_w     = (const float*)d_in[2];  // (DI,1,DC)
  const float* conv_b     = (const float*)d_in[3];  // (DI)
  const float* x_proj_w   = (const float*)d_in[4];  // (33, DI)
  const float* dt_w       = (const float*)d_in[5];  // (DI,1)
  const float* dt_b       = (const float*)d_in[6];  // (DI)
  const float* A_log      = (const float*)d_in[7];  // (DI, NST)
  const float* D_param    = (const float*)d_in[8];  // (DI)
  const float* out_proj_w = (const float*)d_in[9];  // (EDIM, DI)
  float* out = (float*)d_out;                       // (B, L, EDIM) fp32

  char* ws = (char*)d_ws;
  const size_t MB = 1024ull * 1024ull;
  unsigned short* x_bf    = (unsigned short*)(ws + 0);         //  8 MB: (B*L, EDIM)
  unsigned short* win_bf  = (unsigned short*)(ws + 8 * MB);    //  8 MB: (2*DI, EDIM)
  unsigned short* wout_bf = (unsigned short*)(ws + 16 * MB);   //  4 MB: (EDIM, DI)
  float*          xz      = (float*)(ws + 20 * MB);            // 64 MB: (B*L, 2*DI)
  float*          xs      = (float*)(ws + 84 * MB);            // 32 MB: (B*L, DI)
  float*          ssm     = (float*)(ws + 116 * MB);           // ~0.5 MB: (B*L, 33)
  unsigned short* y_bf    = (unsigned short*)(ws + 117 * MB);  // 16 MB: (B*L, DI)

  const long long ML = (long long)BATCH * SEQL;  // 4096

  // 1) fp32 -> bf16 conversions (activations + both big weight matrices)
  k_f2bf<<<dim3(2048), dim3(256), 0, stream>>>(x, x_bf, ML * EDIM);
  k_f2bf<<<dim3(2048), dim3(256), 0, stream>>>(in_proj_w, win_bf,
                                               (long long)2 * DI * EDIM);
  k_f2bf<<<dim3(1024), dim3(256), 0, stream>>>(out_proj_w, wout_bf,
                                               (long long)EDIM * DI);

  // 2) in-projection: xz = x @ in_proj_w^T   (4096 x 4096, K=1024)
  k_gemm_bf16<<<dim3(4096 / 128, 4096 / 256), dim3(256), 0, stream>>>(
      x_bf, win_bf, xz, 4096, 4096, EDIM);

  // 3) depthwise causal conv + silu -> xs (B*L, DI)
  {
    const long long total = ML * DI;
    const int blocks = (int)((total + 255) / 256);
    k_conv_silu<<<dim3(blocks), dim3(256), 0, stream>>>(xz, conv_w, conv_b, xs);
  }

  // 4) ssm projection: ssm = xs @ x_proj_w^T (skinny N=33)
  k_xproj<<<dim3((unsigned)ML), dim3(256), 0, stream>>>(xs, x_proj_w, ssm);

  // 5) selective scan + D skip + silu(z) gating -> y_bf (bf16)
  k_scan<<<dim3(DI / 16, BATCH), dim3(256), 0, stream>>>(
      ssm, xs, xz, dt_w, dt_b, A_log, D_param, y_bf);

  // 6) out-projection: out = y @ out_proj_w^T (4096 x 1024, K=2048)
  k_gemm_bf16<<<dim3(4096 / 128, 1024 / 256), dim3(256), 0, stream>>>(
      y_bf, wout_bf, out, 4096, 1024, DI);
}
